// PartialLinear_49486613185310
// MI455X (gfx1250) — compile-verified, matrix operand order checked by reading, not run on hardware
//
#include <hip/hip_runtime.h>

// ---------------------------------------------------------------------------
// y = x @ (topk_mask(|W|, 1024) * W)^T + bias
//   x: [8,2048,2048] f32 -> M=16384, K=2048 ; W: [2048,2048] f32 (N x K)
//
// Phase 1: exact per-row 1024-th largest |w| via radix select on abs bits.
// Phase 2a (if workspace >= 8KB + 8MB): W masked+converted to f16 ONCE, then
//          GEMM streams B tiles with GLOBAL_LOAD_ASYNC_TO_LDS_B128 (ASYNCcnt,
//          explicit double-buffered LDS, unrolled x2 so all LDS addresses are
//          immediates), converts x f32->f16 in-loop, and runs
//          v_wmma_f32_16x16x32_f16 (f32 accumulate).
// Phase 2b (fallback, tiny workspace): fully fused mask+convert GEMM.
// ---------------------------------------------------------------------------

typedef _Float16 v16h __attribute__((ext_vector_type(16)));
typedef _Float16 v8h  __attribute__((ext_vector_type(8)));
typedef _Float16 v4h  __attribute__((ext_vector_type(4)));
typedef float    v8f  __attribute__((ext_vector_type(8)));
typedef int      v4i_g __attribute__((vector_size(16)));   // matches builtin param

// ---------------- CDNA5 async global->LDS helpers --------------------------
__device__ __forceinline__ void async_ld_b128(const void* g, void* l) {
#if __has_builtin(__builtin_amdgcn_global_load_async_to_lds_b128)
    __builtin_amdgcn_global_load_async_to_lds_b128(
        (__attribute__((address_space(1))) v4i_g*)g,
        (__attribute__((address_space(3))) v4i_g*)l, 0, 0);
#else
    unsigned lo = (unsigned)(size_t)(__attribute__((address_space(3))) void*)l;
    asm volatile("global_load_async_to_lds_b128 %0, %1, off"
                 :: "v"(lo), "v"(g) : "memory");
#endif
}

__device__ __forceinline__ void wait_async0() {
#if __has_builtin(__builtin_amdgcn_s_wait_asynccnt)
    __builtin_amdgcn_s_wait_asynccnt(0);
#else
    asm volatile("s_wait_asynccnt 0" ::: "memory");
#endif
}

#define K_DIM 2048
#define N_DIM 2048
#define M_DIM 16384
#define TOPK  1024

#define BM 128
#define BN 128
#define BK 32
#define LDT 40   // padded LDS row stride in halves (80B) -> conflict-free b128 access

// ---------------------------------------------------------------------------
// Kernel 1: per-row threshold = 1024-th largest abs bit pattern (exact).
// ---------------------------------------------------------------------------
__global__ __launch_bounds__(256) void topk_thresh_kernel(
    const float* __restrict__ W, unsigned* __restrict__ thr)
{
    __shared__ unsigned hist[256];
    __shared__ unsigned s_prefix, s_k;

    const int row = blockIdx.x;
    const int t   = threadIdx.x;

    unsigned key[8];
    const float* wr = W + (size_t)row * K_DIM;
#pragma unroll
    for (int j = 0; j < 8; ++j)
        key[j] = __float_as_uint(wr[t + j * 256]) & 0x7fffffffu;

    unsigned prefix = 0u;
    unsigned k = TOPK;

#pragma unroll
    for (int shift = 24; shift >= 0; shift -= 8) {
        hist[t] = 0u;
        __syncthreads();
        const unsigned hiMask = (shift == 24) ? 0u : (0xffffffffu << (shift + 8));
#pragma unroll
        for (int j = 0; j < 8; ++j) {
            if ((key[j] & hiMask) == prefix)
                atomicAdd(&hist[(key[j] >> shift) & 255u], 1u);
        }
        __syncthreads();
        if (t == 0) {
            unsigned cum = 0u;
            int d = 255;
            for (; d > 0; --d) {
                cum += hist[d];
                if (cum >= k) break;
            }
            if (cum < k) cum += hist[0];
            s_prefix = prefix | ((unsigned)d << shift);
            s_k      = k - (cum - hist[d]);
        }
        __syncthreads();
        prefix = s_prefix;
        k      = s_k;
        __syncthreads();
    }

    if (t == 0) thr[row] = prefix;   // keep iff absbits >= prefix (exactly 1024)
}

// ---------------------------------------------------------------------------
// Kernel 2: one-time W mask + f32->f16 (eliminates 128x redundant conversion)
// ---------------------------------------------------------------------------
__global__ __launch_bounds__(256) void w_prep_kernel(
    const float* __restrict__ W, const unsigned* __restrict__ thr,
    _Float16* __restrict__ Wh)
{
    const int row = blockIdx.x;
    const int t   = threadIdx.x;
    const unsigned T = thr[row];
    const float* wr = W  + (size_t)row * K_DIM;
    _Float16*    wh = Wh + (size_t)row * K_DIM;
#pragma unroll
    for (int s = t; s < K_DIM / 4; s += 256) {
        float4 w4 = *(const float4*)(wr + s * 4);
        w4.x = (__float_as_uint(__builtin_fabsf(w4.x)) >= T) ? w4.x : 0.0f;
        w4.y = (__float_as_uint(__builtin_fabsf(w4.y)) >= T) ? w4.y : 0.0f;
        w4.z = (__float_as_uint(__builtin_fabsf(w4.z)) >= T) ? w4.z : 0.0f;
        w4.w = (__float_as_uint(__builtin_fabsf(w4.w)) >= T) ? w4.w : 0.0f;
        v4h h = { (_Float16)w4.x, (_Float16)w4.y, (_Float16)w4.z, (_Float16)w4.w };
        *(v4h*)(wh + s * 4) = h;
    }
}

// ---------------------------------------------------------------------------
// Kernel 3: GEMM, B tiles via async global->LDS (explicit double buffers),
// x converted in-loop, 8 waves, 128x128 C tile, wave -> 32x64 (2x4 frags).
// ---------------------------------------------------------------------------
__global__ __launch_bounds__(256) void gemm_async_kernel(
    const float* __restrict__ X, const _Float16* __restrict__ Wh,
    const float* __restrict__ bias, float* __restrict__ Y)
{
    __shared__ _Float16 As[BM * LDT];
    __shared__ _Float16 Bs0[BN * LDT];
    __shared__ _Float16 Bs1[BN * LDT];

    const int t  = threadIdx.x;
    const int m0 = blockIdx.y * BM;
    const int n0 = blockIdx.x * BN;

    const int lrow = t >> 3;          // 0..31
    const int lk4  = (t & 7) << 2;    // 0..28 step 4

    const float* xg = X + (size_t)m0 * K_DIM;

    const int lane = t & 31;
    const int wave = t >> 5;
    const int wm = (wave & 3) * 32;
    const int wn = (wave >> 2) * 64;
    const int fr = lane & 15;
    const int kh = (lane >> 4) << 3;  // 0 or 8

    v8f acc[2][4] = {};
    float4 xa[4];

    auto issueB = [&](int kk, _Float16* dst) {
#pragma unroll
        for (int s = 0; s < 2; ++s) {
            const int seg = t + s * 256;          // 512 x 16B segments
            const int row = seg >> 2;
            const int ks  = (seg & 3) << 3;       // k offset in halves
            async_ld_b128(Wh + (size_t)(n0 + row) * K_DIM + kk + ks,
                          dst + row * LDT + ks);
        }
    };
    auto loadX = [&](int kk) {
#pragma unroll
        for (int r = 0; r < 4; ++r)
            xa[r] = *(const float4*)(xg + (size_t)(lrow + r * 32) * K_DIM + kk + lk4);
    };
    auto storeX = [&]() {
#pragma unroll
        for (int r = 0; r < 4; ++r) {
            v4h hx = { (_Float16)xa[r].x, (_Float16)xa[r].y,
                       (_Float16)xa[r].z, (_Float16)xa[r].w };
            *(v4h*)&As[(lrow + r * 32) * LDT + lk4] = hx;
        }
    };
    auto mma = [&](const _Float16* bs) {
        v16h a[2], b[4];
#pragma unroll
        for (int mt = 0; mt < 2; ++mt) {
            const _Float16* p = &As[(wm + mt * 16 + fr) * LDT + kh];
            v8h lo = *(const v8h*)p;
            v8h hi = *(const v8h*)(p + 16);
#pragma unroll
            for (int i = 0; i < 8; ++i) { a[mt][i] = lo[i]; a[mt][i + 8] = hi[i]; }
        }
#pragma unroll
        for (int nt = 0; nt < 4; ++nt) {
            const _Float16* p = &bs[(wn + nt * 16 + fr) * LDT + kh];
            v8h lo = *(const v8h*)p;
            v8h hi = *(const v8h*)(p + 16);
#pragma unroll
            for (int i = 0; i < 8; ++i) { b[nt][i] = lo[i]; b[nt][i + 8] = hi[i]; }
        }
#pragma unroll
        for (int mt = 0; mt < 2; ++mt)
#pragma unroll
            for (int nt = 0; nt < 4; ++nt)
                acc[mt][nt] = __builtin_amdgcn_wmma_f32_16x16x32_f16(
                    false, a[mt], false, b[nt],
                    (short)0, acc[mt][nt], false, false);
    };

    // prologue
    issueB(0, Bs0);
    loadX(0);

#pragma unroll 1
    for (int kk = 0; kk < K_DIM; kk += 2 * BK) {
        // -------- stage 0: compute Bs0, stream kk+BK into Bs1 --------
        storeX();
        wait_async0();
        __syncthreads();
        issueB(kk + BK, Bs1);         // kk+BK <= K_DIM-BK always
        loadX(kk + BK);
        mma(Bs0);
        __syncthreads();

        // -------- stage 1: compute Bs1, stream kk+2*BK into Bs0 --------
        storeX();
        wait_async0();
        __syncthreads();
        if (kk + 2 * BK < K_DIM) {
            issueB(kk + 2 * BK, Bs0);
            loadX(kk + 2 * BK);
        }
        mma(Bs1);
        __syncthreads();
    }

#pragma unroll
    for (int nt = 0; nt < 4; ++nt) {
        const int ncol = n0 + wn + nt * 16 + fr;
        const float bv = bias[ncol];
#pragma unroll
        for (int mt = 0; mt < 2; ++mt) {
            const int mbase = m0 + wm + mt * 16 + kh;
#pragma unroll
            for (int r = 0; r < 8; ++r)
                Y[(size_t)(mbase + r) * N_DIM + ncol] = acc[mt][nt][r] + bv;
        }
    }
}

// ---------------------------------------------------------------------------
// Fallback: fully fused kernel, used when workspace is tiny.
// ---------------------------------------------------------------------------
__global__ __launch_bounds__(256) void gemm_fused_kernel(
    const float* __restrict__ X, const float* __restrict__ W,
    const float* __restrict__ bias, const unsigned* __restrict__ thr,
    float* __restrict__ Y)
{
    __shared__ _Float16 As[BM * LDT];
    __shared__ _Float16 Bs[BN * LDT];

    const int t  = threadIdx.x;
    const int m0 = blockIdx.y * BM;
    const int n0 = blockIdx.x * BN;

    const int lrow = t >> 3;
    const int lk4  = (t & 7) << 2;

    const float* xg = X + (size_t)m0 * K_DIM;
    const float* wg = W + (size_t)n0 * K_DIM;

    unsigned tb[4];
#pragma unroll
    for (int r = 0; r < 4; ++r)
        tb[r] = thr[n0 + lrow + r * 32];

    float4 xa[4], wa[4];

    const int lane = t & 31;
    const int wave = t >> 5;
    const int wm = (wave & 3) * 32;
    const int wn = (wave >> 2) * 64;
    const int fr = lane & 15;
    const int kh = (lane >> 4) << 3;

    v8f acc[2][4] = {};

#pragma unroll
    for (int r = 0; r < 4; ++r) {
        const int rr = lrow + r * 32;
        xa[r] = *(const float4*)(xg + (size_t)rr * K_DIM + lk4);
        wa[r] = *(const float4*)(wg + (size_t)rr * K_DIM + lk4);
    }

    for (int kk = 0; kk < K_DIM; kk += BK) {
#pragma unroll
        for (int r = 0; r < 4; ++r) {
            const int rr = lrow + r * 32;
            v4h hx = { (_Float16)xa[r].x, (_Float16)xa[r].y,
                       (_Float16)xa[r].z, (_Float16)xa[r].w };
            *(v4h*)&As[rr * LDT + lk4] = hx;

            float4 w4 = wa[r];
            const unsigned T = tb[r];
            w4.x = (__float_as_uint(__builtin_fabsf(w4.x)) >= T) ? w4.x : 0.0f;
            w4.y = (__float_as_uint(__builtin_fabsf(w4.y)) >= T) ? w4.y : 0.0f;
            w4.z = (__float_as_uint(__builtin_fabsf(w4.z)) >= T) ? w4.z : 0.0f;
            w4.w = (__float_as_uint(__builtin_fabsf(w4.w)) >= T) ? w4.w : 0.0f;
            v4h hw = { (_Float16)w4.x, (_Float16)w4.y,
                       (_Float16)w4.z, (_Float16)w4.w };
            *(v4h*)&Bs[rr * LDT + lk4] = hw;
        }
        __syncthreads();

        if (kk + BK < K_DIM) {
#pragma unroll
            for (int r = 0; r < 4; ++r) {
                const int rr = lrow + r * 32;
                xa[r] = *(const float4*)(xg + (size_t)rr * K_DIM + kk + BK + lk4);
                wa[r] = *(const float4*)(wg + (size_t)rr * K_DIM + kk + BK + lk4);
            }
        }

        v16h a[2], b[4];
#pragma unroll
        for (int mt = 0; mt < 2; ++mt) {
            const _Float16* p = &As[(wm + mt * 16 + fr) * LDT + kh];
            v8h lo = *(const v8h*)p;
            v8h hi = *(const v8h*)(p + 16);
#pragma unroll
            for (int i = 0; i < 8; ++i) { a[mt][i] = lo[i]; a[mt][i + 8] = hi[i]; }
        }
#pragma unroll
        for (int nt = 0; nt < 4; ++nt) {
            const _Float16* p = &Bs[(wn + nt * 16 + fr) * LDT + kh];
            v8h lo = *(const v8h*)p;
            v8h hi = *(const v8h*)(p + 16);
#pragma unroll
            for (int i = 0; i < 8; ++i) { b[nt][i] = lo[i]; b[nt][i + 8] = hi[i]; }
        }

#pragma unroll
        for (int mt = 0; mt < 2; ++mt)
#pragma unroll
            for (int nt = 0; nt < 4; ++nt)
                acc[mt][nt] = __builtin_amdgcn_wmma_f32_16x16x32_f16(
                    false, a[mt], false, b[nt],
                    (short)0, acc[mt][nt], false, false);

        __syncthreads();
    }

#pragma unroll
    for (int nt = 0; nt < 4; ++nt) {
        const int ncol = n0 + wn + nt * 16 + fr;
        const float bv = bias[ncol];
#pragma unroll
        for (int mt = 0; mt < 2; ++mt) {
            const int mbase = m0 + wm + mt * 16 + kh;
#pragma unroll
            for (int r = 0; r < 8; ++r)
                Y[(size_t)(mbase + r) * N_DIM + ncol] = acc[mt][nt][r] + bv;
        }
    }
}

// ---------------------------------------------------------------------------
extern "C" void kernel_launch(void* const* d_in, const int* in_sizes, int n_in,
                              void* d_out, int out_size, void* d_ws, size_t ws_size,
                              hipStream_t stream)
{
    const float* x    = (const float*)d_in[0];   // [8,2048,2048]
    const float* w    = (const float*)d_in[1];   // [2048,2048]
    const float* bias = (const float*)d_in[2];   // [2048]
    float*       y    = (float*)d_out;

    unsigned* thr = (unsigned*)d_ws;                               // 8 KB
    const size_t thr_bytes = (size_t)N_DIM * sizeof(unsigned);
    const size_t wh_bytes  = (size_t)N_DIM * K_DIM * sizeof(_Float16);  // 8 MB

    topk_thresh_kernel<<<dim3(N_DIM), dim3(256), 0, stream>>>(w, thr);

    if (ws_size >= thr_bytes + wh_bytes) {
        _Float16* wh = (_Float16*)((char*)d_ws + thr_bytes);
        w_prep_kernel<<<dim3(N_DIM), dim3(256), 0, stream>>>(w, thr, wh);
        gemm_async_kernel<<<dim3(N_DIM / BN, M_DIM / BM), dim3(256), 0, stream>>>(
            x, wh, bias, y);
    } else {
        gemm_fused_kernel<<<dim3(N_DIM / BN, M_DIM / BM), dim3(256), 0, stream>>>(
            x, w, bias, thr, y);
    }
}
